// MambaController_38336878084578
// MI455X (gfx1250) — matrix operand order repaired and checked by reading, not compiled, not run on hardware
//
#include <hip/hip_runtime.h>

#define BBATCH 2
#define LSEQ   2048
#define NTOK   4096      // B*L
#define OBS_D  256
#define ACT_D  64
#define DMODEL 1024
#define DINNER 2048
#define DSTATE 16
#define DCONV  4
#define DTRANK 64
#define DBC    96        // DT_RANK + 2*D_STATE
#define NCH    16        // scan chunks
#define CHLEN  128       // LSEQ / NCH

// LDS-tiled GEMM geometry
#define BM 64
#define BN 128
#define BK 32
#define LST 40           // BK + 8 pad elems (80B row stride -> conflict-free frags)
// dynamic-LDS byte offsets (row = 64B data + 16B pad = 80B)
#define SA_BYTES (BM * LST * 2)              // 5120
#define SB_BYTES (BN * LST * 2)              // 10240
#define SA_OFF(buf) ((unsigned)((buf) * SA_BYTES))
#define SB_OFF(buf) ((unsigned)(2 * SA_BYTES + (buf) * SB_BYTES))
#define SMEM_BYTES (2 * SA_BYTES + 2 * SB_BYTES)   // 30720

typedef __attribute__((ext_vector_type(8)))  __bf16 bf16x8;
typedef __attribute__((ext_vector_type(16))) __bf16 bf16x16;
typedef __attribute__((ext_vector_type(8)))  float  f32x8;
typedef unsigned int u32;
typedef __attribute__((ext_vector_type(4))) u32 u32x4;
typedef __attribute__((ext_vector_type(4))) int i32x4;
typedef __attribute__((ext_vector_type(8))) int i32x8;

#if defined(__gfx1250__) && __has_builtin(__builtin_amdgcn_tensor_load_to_lds)
#define HAVE_TDM 1
#else
#define HAVE_TDM 0
#endif

__device__ __forceinline__ float silu_(float x) { return x / (1.f + __expf(-x)); }
__device__ __forceinline__ bf16x16 cat16(bf16x8 lo, bf16x8 hi) {
  return __builtin_shufflevector(lo, hi, 0,1,2,3,4,5,6,7,8,9,10,11,12,13,14,15);
}

// Issue one TDM 2-D tile load (bf16 elements) with LDS row padding:
// tile0 x tile1 elements, global row stride st0 (elems), pad 16B per 64B row.
__device__ __forceinline__ void tdm_load_tile(
    const void* gaddr, unsigned lds_off,
    unsigned td0, unsigned td1, unsigned tile0, unsigned tile1, unsigned st0)
{
#if HAVE_TDM
  unsigned long long ga = (unsigned long long)gaddr;
  u32x4 g0;
  g0[0] = 1u;                                   // count=1, user mode
  g0[1] = lds_off;                              // lds_addr (bytes)
  g0[2] = (u32)ga;                              // global_addr[31:0]
  g0[3] = (u32)((ga >> 32) & 0x1FFFFFFu) | 0x80000000u;  // addr[56:32] | type=2
  i32x8 g1;
  // data_size=2B (code1), pad_enable, pad_interval=16DW(code3), pad_amount=4DW(code3)
  g1[0] = (int)((1u << 16) | (1u << 20) | (3u << 22) | (3u << 25));
  g1[1] = (int)((td0 & 0xFFFFu) << 16);                       // tensor_dim0 lo
  g1[2] = (int)(((td0 >> 16) & 0xFFFFu) | ((td1 & 0xFFFFu) << 16));
  g1[3] = (int)(((td1 >> 16) & 0xFFFFu) | ((tile0 & 0xFFFFu) << 16));
  g1[4] = (int)(tile1 & 0xFFFFu);                             // tile_dim1 (tile_dim2=0)
  g1[5] = (int)st0;                                           // dim0 stride lo
  g1[6] = 0;                                                  // stride hi / dim1 stride
  g1[7] = 0;
  i32x4 gz = {};
#if defined(__clang_major__) && __clang_major__ >= 23
  i32x8 gz8 = {};
  __builtin_amdgcn_tensor_load_to_lds(g0, g1, gz, gz, gz8, 0);
#else
  __builtin_amdgcn_tensor_load_to_lds(g0, g1, gz, gz, 0);
#endif
#else
  (void)gaddr; (void)lds_off; (void)td0; (void)td1; (void)tile0; (void)tile1; (void)st0;
#endif
}

// fp32 -> bf16, 8 elements/thread
__global__ __launch_bounds__(256) void k_cvt_bf16(
    const float* __restrict__ s, __bf16* __restrict__ d, int n8)
{
  int i = blockIdx.x * 256 + threadIdx.x;
  if (i >= n8) return;
  float4 a = ((const float4*)s)[i * 2];
  float4 b = ((const float4*)s)[i * 2 + 1];
  bf16x8 v;
  v[0]=(__bf16)a.x; v[1]=(__bf16)a.y; v[2]=(__bf16)a.z; v[3]=(__bf16)a.w;
  v[4]=(__bf16)b.x; v[5]=(__bf16)b.y; v[6]=(__bf16)b.z; v[7]=(__bf16)b.w;
  ((bf16x8*)d)[i] = v;
}

// ---------------- LDS-tiled WMMA GEMM with TDM staging ----------------
// C[M,N] = act( A[M,K(lda)] @ W[N,K]^T + bias ), M=NTOK, N%128==0, K%32==0.
// 256 threads = 8 waves as 2(M) x 4(N); each wave: 2x2 wmma tiles (32x32).
// Wave 0 drives the Tensor Data Mover into double-buffered LDS.
template<int ACT, int WBF>
__global__ __launch_bounds__(256) void k_gemm_lds(
    const __bf16* __restrict__ A, int lda,
    const __bf16* __restrict__ W,
    const float* __restrict__ bias,
    float* __restrict__ C, __bf16* __restrict__ Cb,
    int N, int K)
{
  extern __shared__ char smem[];
  const int tid  = threadIdx.x;
  const int lane = tid & 31, wave = tid >> 5;
  const int wm   = wave >> 2, wn = wave & 3;
  const int half = lane >> 4, sub = lane & 15;
  const int m0   = blockIdx.y * BM, n0 = blockIdx.x * BN;

  const __bf16* Abase = A + (size_t)m0 * lda;
  const __bf16* Bbase = W + (size_t)n0 * K;

#if HAVE_TDM
  if (wave == 0) {
    tdm_load_tile(Abase, SA_OFF(0), (unsigned)lda, 1u << 20, BK, BM, (unsigned)lda);
    tdm_load_tile(Bbase, SB_OFF(0), (unsigned)K,   1u << 20, BK, BN, (unsigned)K);
    __builtin_amdgcn_s_wait_tensorcnt(0);
  }
#else
  {
    const int ar = tid >> 2, ac = (tid & 3) << 3;   // A tile: 64 rows x 4 chunks(8)
    const int br = tid >> 1, bc = (tid & 1) << 4;   // B tile: 128 rows x 2 chunks(16)
    bf16x8 ga  = *(const bf16x8*)(Abase + (size_t)ar * lda + ac);
    bf16x8 gb0 = *(const bf16x8*)(Bbase + (size_t)br * K + bc);
    bf16x8 gb1 = *(const bf16x8*)(Bbase + (size_t)br * K + bc + 8);
    *(bf16x8*)(smem + SA_OFF(0) + (ar * LST + ac) * 2)     = ga;
    *(bf16x8*)(smem + SB_OFF(0) + (br * LST + bc) * 2)     = gb0;
    *(bf16x8*)(smem + SB_OFF(0) + (br * LST + bc + 8) * 2) = gb1;
  }
#endif
  __syncthreads();

  f32x8 acc[2][2];
#pragma unroll
  for (int i = 0; i < 2; ++i)
#pragma unroll
    for (int j = 0; j < 2; ++j) acc[i][j] = (f32x8){};

  int buf = 0;
  for (int k0 = 0; k0 < K; k0 += BK) {
    const int kn = k0 + BK;
#if HAVE_TDM
    if (wave == 0 && kn < K) {
      tdm_load_tile(Abase + kn, SA_OFF(buf ^ 1), (unsigned)lda, 1u << 20, BK, BM, (unsigned)lda);
      tdm_load_tile(Bbase + kn, SB_OFF(buf ^ 1), (unsigned)K,   1u << 20, BK, BN, (unsigned)K);
    }
#else
    bf16x8 na = {}, nb0 = {}, nb1 = {};
    const int ar = tid >> 2, ac = (tid & 3) << 3;
    const int br = tid >> 1, bc = (tid & 1) << 4;
    if (kn < K) {
      na  = *(const bf16x8*)(Abase + (size_t)ar * lda + ac + kn);
      nb0 = *(const bf16x8*)(Bbase + (size_t)br * K + bc + kn);
      nb1 = *(const bf16x8*)(Bbase + (size_t)br * K + bc + 8 + kn);
    }
#endif
    // fragments from LDS
    bf16x16 af[2], bfr[2];
#pragma unroll
    for (int i = 0; i < 2; ++i) {
      const int row = wm * 32 + i * 16 + sub;
      bf16x8 lo = *(const bf16x8*)(smem + SA_OFF(buf) + (row * LST + half * 8) * 2);
      bf16x8 hi = *(const bf16x8*)(smem + SA_OFF(buf) + (row * LST + 16 + half * 8) * 2);
      af[i] = cat16(lo, hi);
      const int col = wn * 32 + i * 16 + sub;
      bf16x8 blo = *(const bf16x8*)(smem + SB_OFF(buf) + (col * LST + half * 8) * 2);
      bf16x8 bhi = *(const bf16x8*)(smem + SB_OFF(buf) + (col * LST + 16 + half * 8) * 2);
      bfr[i] = cat16(blo, bhi);
    }
#pragma unroll
    for (int i = 0; i < 2; ++i)
#pragma unroll
      for (int j = 0; j < 2; ++j)
        acc[i][j] = __builtin_amdgcn_wmma_f32_16x16x32_bf16(
            false, af[i], false, bfr[j], (short)0, acc[i][j], false, false);

#if HAVE_TDM
    if (wave == 0 && kn < K) __builtin_amdgcn_s_wait_tensorcnt(0);
#else
    if (kn < K) {
      *(bf16x8*)(smem + SA_OFF(buf ^ 1) + (ar * LST + ac) * 2)     = na;
      *(bf16x8*)(smem + SB_OFF(buf ^ 1) + (br * LST + bc) * 2)     = nb0;
      *(bf16x8*)(smem + SB_OFF(buf ^ 1) + (br * LST + bc + 8) * 2) = nb1;
    }
#endif
    __syncthreads();
    buf ^= 1;
  }

#pragma unroll
  for (int i = 0; i < 2; ++i) {
    const int rb = m0 + wm * 32 + i * 16 + half * 8;
#pragma unroll
    for (int j = 0; j < 2; ++j) {
      const int col = n0 + wn * 32 + j * 16 + sub;
      const float bv = bias ? bias[col] : 0.f;
#pragma unroll
      for (int r = 0; r < 8; ++r) {
        float v = acc[i][j][r] + bv;
        if (ACT == 1) v = (v > 20.f) ? v : log1pf(__expf(v));  // softplus
        C[(size_t)(rb + r) * N + col] = v;
        if (WBF) Cb[(size_t)(rb + r) * N + col] = (__bf16)v;
      }
    }
  }
}

// ---------------- direct WMMA GEMM (odd N: 96, 64) ----------------
template<int WBF, int TN>
__global__ __launch_bounds__(256) void k_gemm_direct(
    const __bf16* __restrict__ A, int lda,
    const __bf16* __restrict__ W,
    const float* __restrict__ bias,
    float* __restrict__ C, __bf16* __restrict__ Cb, int N, int K)
{
  const int lane = threadIdx.x & 31;
  const int wave = threadIdx.x >> 5;
  const int tnb  = blockIdx.x * TN;
  const int tm   = blockIdx.y * 8 + wave;
  const int half = lane >> 4, sub = lane & 15;

  const __bf16* ap = A + (size_t)(tm * 16 + sub) * lda + half * 8;
  const __bf16* bp[TN];
#pragma unroll
  for (int nt = 0; nt < TN; ++nt)
    bp[nt] = W + (size_t)((tnb + nt) * 16 + sub) * K + half * 8;

  f32x8 acc[TN];
#pragma unroll
  for (int nt = 0; nt < TN; ++nt) acc[nt] = (f32x8){};

  for (int k0 = 0; k0 < K; k0 += 32) {
    __builtin_prefetch(ap + k0 + 32, 0, 0);
    bf16x16 a = cat16(*(const bf16x8*)(ap + k0), *(const bf16x8*)(ap + k0 + 16));
#pragma unroll
    for (int nt = 0; nt < TN; ++nt) {
      __builtin_prefetch(bp[nt] + k0 + 32, 0, 0);
      bf16x16 b = cat16(*(const bf16x8*)(bp[nt] + k0), *(const bf16x8*)(bp[nt] + k0 + 16));
      acc[nt] = __builtin_amdgcn_wmma_f32_16x16x32_bf16(
          false, a, false, b, (short)0, acc[nt], false, false);
    }
  }

  const int rb = tm * 16 + half * 8;
#pragma unroll
  for (int nt = 0; nt < TN; ++nt) {
    const int col = (tnb + nt) * 16 + sub;
    const float bv = bias ? bias[col] : 0.f;
#pragma unroll
    for (int r = 0; r < 8; ++r) {
      float v = acc[nt][r] + bv;
      C[(size_t)(rb + r) * N + col] = v;
      if (WBF) Cb[(size_t)(rb + r) * N + col] = (__bf16)v;
    }
  }
}

// causal depthwise conv (k=4) + bias + SiLU; writes fp32 + bf16 copies
__global__ __launch_bounds__(256) void k_conv_silu(
    const float* __restrict__ xz, const float* __restrict__ cw,
    const float* __restrict__ cb, float* __restrict__ xc, __bf16* __restrict__ xcb)
{
  int idx = blockIdx.x * 256 + threadIdx.x;   // [T, DINNER]
  int c  = idx & (DINNER - 1);
  int t  = idx >> 11;
  int tl = t & (LSEQ - 1);
  float acc = cb[c];
#pragma unroll
  for (int i = 0; i < DCONV; ++i) {
    int s = tl - (DCONV - 1) + i;
    if (s >= 0)
      acc += xz[(size_t)(t - (DCONV - 1) + i) * (2 * DINNER) + c] * cw[c * DCONV + i];
  }
  float v = silu_(acc);
  xc[idx]  = v;
  xcb[idx] = (__bf16)v;
}

__global__ __launch_bounds__(256) void k_negexpA(
    const float* __restrict__ Alog, float* __restrict__ Aout)
{
  int i = blockIdx.x * 256 + threadIdx.x;
  Aout[i] = -__expf(Alog[i]);
}

// Pass 1: per-(b,chunk,d) local scan from h=0; emit chunk state + dA product.
__global__ __launch_bounds__(256) void k_scan_local(
    const float* __restrict__ delta, const float* __restrict__ xc,
    const float* __restrict__ dbc, const float* __restrict__ Aneg,
    float* __restrict__ hloc, float* __restrict__ Pprod)
{
  int idx = blockIdx.x * 256 + threadIdx.x;   // [B, NCH, DINNER]
  int d  = idx & (DINNER - 1);
  int ch = (idx >> 11) & (NCH - 1);
  int b  = idx >> 15;
  float a[DSTATE], h[DSTATE], p[DSTATE];
#pragma unroll
  for (int n = 0; n < DSTATE; ++n) { a[n] = Aneg[d * DSTATE + n]; h[n] = 0.f; p[n] = 1.f; }
  int t0 = b * LSEQ + ch * CHLEN;
  for (int t = 0; t < CHLEN; ++t) {
    int gt = t0 + t;
    float de = delta[(size_t)gt * DINNER + d];
    float dx = de * xc[(size_t)gt * DINNER + d];
    const float* bcp = dbc + (size_t)gt * DBC;
#pragma unroll
    for (int n = 0; n < DSTATE; ++n) {
      float dA = __expf(de * a[n]);
      h[n] = dA * h[n] + dx * bcp[DTRANK + n];
      p[n] *= dA;
    }
  }
  size_t base = (size_t)idx * DSTATE;
#pragma unroll
  for (int n = 0; n < DSTATE; ++n) { hloc[base + n] = h[n]; Pprod[base + n] = p[n]; }
}

// Pass 2: sequential chunk-prefix per (b,d); rewrites hloc with incoming states.
__global__ __launch_bounds__(256) void k_scan_combine(
    float* __restrict__ hloc, const float* __restrict__ Pprod)
{
  int idx = blockIdx.x * 256 + threadIdx.x;   // [B, DINNER]
  int d = idx & (DINNER - 1);
  int b = idx >> 11;
  float h[DSTATE];
#pragma unroll
  for (int n = 0; n < DSTATE; ++n) h[n] = 0.f;
  for (int ch = 0; ch < NCH; ++ch) {
    size_t base = ((size_t)((b * NCH + ch) * DINNER + d)) * DSTATE;
#pragma unroll
    for (int n = 0; n < DSTATE; ++n) {
      float pl = Pprod[base + n];
      float hl = hloc[base + n];
      hloc[base + n] = h[n];            // becomes h_in for chunk ch
      h[n] = pl * h[n] + hl;
    }
  }
}

// Pass 3: replay each chunk from h_in; y = (scan + D*xc) * silu(z), bf16 out.
__global__ __launch_bounds__(256) void k_scan_final(
    const float* __restrict__ delta, const float* __restrict__ xc,
    const float* __restrict__ dbc, const float* __restrict__ Aneg,
    const float* __restrict__ hin, const float* __restrict__ xz,
    const float* __restrict__ Dvec, __bf16* __restrict__ yb)
{
  int idx = blockIdx.x * 256 + threadIdx.x;   // [B, NCH, DINNER]
  int d  = idx & (DINNER - 1);
  int ch = (idx >> 11) & (NCH - 1);
  int b  = idx >> 15;
  float a[DSTATE], h[DSTATE];
  size_t base = (size_t)idx * DSTATE;
#pragma unroll
  for (int n = 0; n < DSTATE; ++n) { a[n] = Aneg[d * DSTATE + n]; h[n] = hin[base + n]; }
  float Dd = Dvec[d];
  int t0 = b * LSEQ + ch * CHLEN;
  for (int t = 0; t < CHLEN; ++t) {
    int gt = t0 + t;
    float de = delta[(size_t)gt * DINNER + d];
    float xv = xc[(size_t)gt * DINNER + d];
    float dx = de * xv;
    const float* bcp = dbc + (size_t)gt * DBC;
    float yt = 0.f;
#pragma unroll
    for (int n = 0; n < DSTATE; ++n) {
      float dA = __expf(de * a[n]);
      h[n] = dA * h[n] + dx * bcp[DTRANK + n];
      yt += h[n] * bcp[DTRANK + DSTATE + n];
    }
    float z = xz[(size_t)gt * (2 * DINNER) + DINNER + d];
    yb[(size_t)gt * DINNER + d] = (__bf16)((yt + Dd * xv) * silu_(z));
  }
}

extern "C" void kernel_launch(void* const* d_in, const int* in_sizes, int n_in,
                              void* d_out, int out_size, void* d_ws, size_t ws_size,
                              hipStream_t stream)
{
  (void)in_sizes; (void)n_in; (void)out_size; (void)ws_size;
  const float* obs        = (const float*)d_in[0];
  const float* in_w       = (const float*)d_in[1];
  const float* in_b       = (const float*)d_in[2];
  const float* out_w      = (const float*)d_in[3];
  const float* out_b      = (const float*)d_in[4];
  const float* in_proj_w  = (const float*)d_in[5];
  const float* conv_w     = (const float*)d_in[6];
  const float* conv_b     = (const float*)d_in[7];
  const float* x_proj_w   = (const float*)d_in[8];
  const float* dt_proj_w  = (const float*)d_in[9];
  const float* dt_proj_b  = (const float*)d_in[10];
  const float* A_log      = (const float*)d_in[11];
  const float* Dp         = (const float*)d_in[12];
  const float* out_proj_w = (const float*)d_in[13];
  float* out = (float*)d_out;

  char* wp = (char*)d_ws;
  auto alloc_f = [&](size_t n) {
    float* p = (float*)wp; wp += ((n * sizeof(float) + 255) & ~(size_t)255); return p;
  };
  auto alloc_h = [&](size_t n) {
    __bf16* p = (__bf16*)wp; wp += ((n * sizeof(__bf16) + 255) & ~(size_t)255); return p;
  };
  float*  x      = alloc_f((size_t)NTOK * DMODEL);
  float*  xz     = alloc_f((size_t)NTOK * 2 * DINNER);
  float*  xc     = alloc_f((size_t)NTOK * DINNER);
  float*  dbc    = alloc_f((size_t)NTOK * DBC);
  float*  delta  = alloc_f((size_t)NTOK * DINNER);
  float*  Aneg   = alloc_f((size_t)DINNER * DSTATE);
  float*  hloc   = alloc_f((size_t)BBATCH * NCH * DINNER * DSTATE);
  float*  Pprod  = alloc_f((size_t)BBATCH * NCH * DINNER * DSTATE);
  __bf16* obs_b  = alloc_h((size_t)NTOK * OBS_D);
  __bf16* x_b    = alloc_h((size_t)NTOK * DMODEL);
  __bf16* xc_b   = alloc_h((size_t)NTOK * DINNER);
  __bf16* dbc_b  = alloc_h((size_t)NTOK * DBC);
  __bf16* y_b    = alloc_h((size_t)NTOK * DINNER);
  __bf16* wbuf   = alloc_h((size_t)2 * DINNER * DMODEL);   // largest weight

  auto cvt = [&](const float* s, __bf16* d, size_t n) {
    int n8 = (int)(n / 8);
    k_cvt_bf16<<<(n8 + 255) / 256, 256, 0, stream>>>(s, d, n8);
  };

  // x = obs @ in_w^T + in_b   (also emits x_b)
  cvt(obs, obs_b, (size_t)NTOK * OBS_D);
  cvt(in_w, wbuf, (size_t)DMODEL * OBS_D);
  k_gemm_lds<0, 1><<<dim3(DMODEL / BN, NTOK / BM), 256, SMEM_BYTES, stream>>>(
      obs_b, OBS_D, wbuf, in_b, x, x_b, DMODEL, OBS_D);

  for (int l = 0; l < 2; ++l) {
    const float* ipw = in_proj_w  + (size_t)l * 2 * DINNER * DMODEL;
    const float* cw  = conv_w     + (size_t)l * DINNER * DCONV;
    const float* cb  = conv_b     + (size_t)l * DINNER;
    const float* xpw = x_proj_w   + (size_t)l * DBC * DINNER;
    const float* dpw = dt_proj_w  + (size_t)l * DINNER * DTRANK;
    const float* dpb = dt_proj_b  + (size_t)l * DINNER;
    const float* alg = A_log      + (size_t)l * DINNER * DSTATE;
    const float* dvl = Dp         + (size_t)l * DINNER;
    const float* opw = out_proj_w + (size_t)l * DMODEL * DINNER;

    // xz = x @ in_proj^T
    cvt(ipw, wbuf, (size_t)2 * DINNER * DMODEL);
    k_gemm_lds<0, 0><<<dim3((2 * DINNER) / BN, NTOK / BM), 256, SMEM_BYTES, stream>>>(
        x_b, DMODEL, wbuf, nullptr, xz, nullptr, 2 * DINNER, DMODEL);

    // xc = silu(conv(xh) + cb)
    k_conv_silu<<<(NTOK * DINNER) / 256, 256, 0, stream>>>(xz, cw, cb, xc, xc_b);

    // dbc = xc @ x_proj^T   (N=96 -> direct kernel; also emits dbc_b)
    cvt(xpw, wbuf, (size_t)DBC * DINNER);
    k_gemm_direct<1, 1><<<dim3(DBC / 16, NTOK / 128), 256, 0, stream>>>(
        xc_b, DINNER, wbuf, nullptr, dbc, dbc_b, DBC, DINNER);

    // delta = softplus(dt @ dt_proj^T + dt_b); dt = dbc[:, :64] (lda=96)
    cvt(dpw, wbuf, (size_t)DINNER * DTRANK);
    k_gemm_lds<1, 0><<<dim3(DINNER / BN, NTOK / BM), 256, SMEM_BYTES, stream>>>(
        dbc_b, DBC, wbuf, dpb, delta, nullptr, DINNER, DTRANK);

    // selective scan (chunked 3-pass)
    k_negexpA<<<(DINNER * DSTATE) / 256, 256, 0, stream>>>(alg, Aneg);
    k_scan_local<<<(BBATCH * NCH * DINNER) / 256, 256, 0, stream>>>(
        delta, xc, dbc, Aneg, hloc, Pprod);
    k_scan_combine<<<(BBATCH * DINNER) / 256, 256, 0, stream>>>(hloc, Pprod);
    k_scan_final<<<(BBATCH * NCH * DINNER) / 256, 256, 0, stream>>>(
        delta, xc, dbc, Aneg, hloc, xz, dvl, y_b);

    // x = y @ out_proj^T   (also emits x_b)
    cvt(opw, wbuf, (size_t)DMODEL * DINNER);
    k_gemm_lds<0, 1><<<dim3(DMODEL / BN, NTOK / BM), 256, SMEM_BYTES, stream>>>(
        y_b, DINNER, wbuf, nullptr, x, x_b, DMODEL, DINNER);
  }

  // out = x @ out_w^T + out_b   (N=64 -> direct kernel)
  cvt(out_w, wbuf, (size_t)ACT_D * DMODEL);
  k_gemm_direct<0, 4><<<dim3(ACT_D / 64, NTOK / 128), 256, 0, stream>>>(
      x_b, DMODEL, wbuf, out_b, out, nullptr, ACT_D, DMODEL);
}
